// SpatialAttention_51436528336963
// MI455X (gfx1250) — matrix operand order repaired and checked by reading, not compiled, not run on hardware
//
#include <hip/hip_runtime.h>

// ---------------------------------------------------------------------------
// SpatialAttention forward for MI455X (gfx1250, wave32, WMMA f16->f32).
// B=4, N=2048, C=512, H=8, D=64.
// ---------------------------------------------------------------------------

typedef __attribute__((ext_vector_type(16))) _Float16 v16h;
typedef __attribute__((ext_vector_type(8)))  _Float16 v8h;
typedef __attribute__((ext_vector_type(8)))  float    v8f;

#define C_DIM   512
#define N_TOK   2048
#define B_DIM   4
#define HEADS   8
#define HEAD_D  64
#define ROWS    (B_DIM * N_TOK)      // 8192
#define QKV_DIM (3 * C_DIM)          // 1536

// ---- WMMA wrapper (codegen-verified builtin signature) --------------------
__device__ static inline v8f wmma16x16x32(v16h a, v16h b, v8f c) {
  return __builtin_amdgcn_wmma_f32_16x16x32_f16(
      /*neg_a=*/false, a, /*neg_b=*/false, b,
      /*c_mod=*/(short)0, c, /*reuse_a=*/false, /*reuse_b=*/false);
}

__device__ static inline v16h pack16(v8h lo, v8h hi) {
  v16h r;
#pragma unroll
  for (int i = 0; i < 8; ++i) { r[i] = lo[i]; r[i + 8] = hi[i]; }
  return r;
}

// A-matrix fragment (16xK slice, contiguous k order, same as load_bt_frag):
// lane L: row m = m0 + (L&15); elements 0..15 <- K = k0 + (L>>4)*16 + {0..15}.
__device__ static inline v16h load_a_frag(const _Float16* __restrict__ base,
                                          int ld, int m0, int k0, int lane) {
  const int m = m0 + (lane & 15);
  const int half = lane >> 4;
  const _Float16* p = base + (size_t)m * ld + k0 + half * 16;
  v8h lo = *(const v8h*)(p);
  v8h hi = *(const v8h*)(p + 8);
  return pack16(lo, hi);
}

// B-matrix fragment where B^T is stored row-major (row = output col j,
// contiguous over k): lane L: col n = n0 + (L&15);
// elements 0..15 <- K = k0 + (L>>4)*16 + {0..15}  (contiguous in memory).
__device__ static inline v16h load_bt_frag(const _Float16* __restrict__ base,
                                           int ld, int n0, int k0, int lane) {
  const int n = n0 + (lane & 15);
  const int half = lane >> 4;
  const _Float16* p = base + (size_t)n * ld + k0 + half * 16;
  v8h lo = *(const v8h*)(p);
  v8h hi = *(const v8h*)(p + 8);
  return pack16(lo, hi);
}

// ---------------------------------------------------------------------------
// Kernel 1: LayerNorm (fp32 in) -> f16 h. One block per row.
// ---------------------------------------------------------------------------
__global__ __launch_bounds__(256)
void sa_ln_kernel(const float* __restrict__ x, const float* __restrict__ g,
                  const float* __restrict__ b, _Float16* __restrict__ h) {
  __shared__ float s1[256];
  __shared__ float s2[256];
  const int row = blockIdx.x;
  const int t = threadIdx.x;
  const float* xr = x + (size_t)row * C_DIM;
  const float a = xr[t];
  const float c = xr[t + 256];
  s1[t] = a + c;
  s2[t] = a * a + c * c;
  __syncthreads();
#pragma unroll
  for (int off = 128; off > 0; off >>= 1) {
    if (t < off) { s1[t] += s1[t + off]; s2[t] += s2[t + off]; }
    __syncthreads();
  }
  const float mu = s1[0] * (1.0f / C_DIM);
  const float var = s2[0] * (1.0f / C_DIM) - mu * mu;
  const float rs = rsqrtf(var + 1e-5f);
  _Float16* hr = h + (size_t)row * C_DIM;
  hr[t]       = (_Float16)((a - mu) * rs * g[t] + b[t]);
  hr[t + 256] = (_Float16)((c - mu) * rs * g[t + 256] + b[t + 256]);
}

// ---------------------------------------------------------------------------
// Kernel 2: fp32 -> f16 conversion (weights).
// ---------------------------------------------------------------------------
__global__ void sa_cvt_f16_kernel(const float* __restrict__ in,
                                  _Float16* __restrict__ out, int n) {
  const int i = blockIdx.x * blockDim.x + threadIdx.x;
  if (i < n) out[i] = (_Float16)in[i];
}

// ---------------------------------------------------------------------------
// Kernel 3: QKV GEMM: qkv[t, j] = sum_c h[t,c] * qkv_w[j,c] + qkv_b[j].
// One wave computes a 32x64 output tile: 2 A-fragments, 8 accumulators,
// each B fragment feeds 2 independent WMMAs (hides load latency, halves
// B-side L2 traffic). Results scattered to q/k/v in [B,H,N,D] f16.
// ---------------------------------------------------------------------------
__global__ __launch_bounds__(256)
void sa_qkv_gemm_kernel(const _Float16* __restrict__ h,
                        const _Float16* __restrict__ w,   // [1536][512] f16
                        const float* __restrict__ qkv_b,
                        _Float16* __restrict__ q, _Float16* __restrict__ k,
                        _Float16* __restrict__ v) {
  const int wave = (blockIdx.x * blockDim.x + threadIdx.x) >> 5;
  const int lane = threadIdx.x & 31;
  const int nb = wave % (QKV_DIM / 64);        // 24 N-blocks of 64
  const int mt = wave / (QKV_DIM / 64);        // 256 M-tiles of 32
  const int m0 = mt * 32;
  const int n0 = nb * 64;

  v8f acc[2][4] = {};
  for (int k0 = 0; k0 < C_DIM; k0 += 32) {
    const v16h a0 = load_a_frag(h, C_DIM, m0, k0, lane);
    const v16h a1 = load_a_frag(h, C_DIM, m0 + 16, k0, lane);
#pragma unroll
    for (int t = 0; t < 4; ++t) {
      const v16h bf = load_bt_frag(w, C_DIM, n0 + t * 16, k0, lane);
      acc[0][t] = wmma16x16x32(a0, bf, acc[0][t]);
      acc[1][t] = wmma16x16x32(a1, bf, acc[1][t]);
    }
  }

  const int half = lane >> 4;
  const int nl = lane & 15;
#pragma unroll
  for (int t = 0; t < 4; ++t) {
    const int j = n0 + t * 16 + nl;
    const int which = j >> 9;          // 0=q, 1=k, 2=v
    const int c2 = j & (C_DIM - 1);
    const int head = c2 >> 6;
    const int d = c2 & (HEAD_D - 1);
    _Float16* dst = (which == 0) ? q : (which == 1) ? k : v;
    const float bj = qkv_b[j];
#pragma unroll
    for (int rt = 0; rt < 2; ++rt) {
#pragma unroll
      for (int r = 0; r < 8; ++r) {
        const int tok = m0 + rt * 16 + r + half * 8;
        const int bi = tok >> 11;                // token / 2048
        const int n = tok & (N_TOK - 1);
        dst[(((size_t)(bi * HEADS + head) * N_TOK) + n) * HEAD_D + d] =
            (_Float16)(acc[rt][t][r] + bj);
      }
    }
  }
}

// ---------------------------------------------------------------------------
// Kernel 4: fused attention (flash-style), 64-key slabs, double-buffered LDS.
// K tiles move via gfx1250 async copies (global_load_async_to_lds_b128,
// ASYNCcnt-pipelined one slab ahead); V tiles are prefetched into registers
// one slab ahead and stored transposed. 16 WMMAs per barrier pair.
// ---------------------------------------------------------------------------
#define SLAB 64
#define NSLAB (N_TOK / SLAB)   // 32

// Per-wave share of one 8KB slab: 2 x (32 lanes x 16B) async copies.
__device__ static inline void async_copy_slab(unsigned lds_base,
                                              const _Float16* gbase,
                                              int wave, int lane) {
  const unsigned off0 = (unsigned)((wave * 64 + lane) * 16);
  const unsigned off1 = off0 + 32 * 16;
  const unsigned long long g = (unsigned long long)(uintptr_t)gbase;
  const unsigned long long ga0 = g + off0;
  const unsigned long long ga1 = g + off1;
  const unsigned la0 = lds_base + off0;
  const unsigned la1 = lds_base + off1;
  asm volatile("global_load_async_to_lds_b128 %0, %1, off"
               :: "v"(la0), "v"(ga0) : "memory");
  asm volatile("global_load_async_to_lds_b128 %0, %1, off"
               :: "v"(la1), "v"(ga1) : "memory");
}

__global__ __launch_bounds__(256)
void sa_attn_kernel(const _Float16* __restrict__ qg,
                    const _Float16* __restrict__ kg,
                    const _Float16* __restrict__ vg,
                    const float* __restrict__ biasg,   // [B,1,N,N] fp32
                    const float* __restrict__ beta_p,
                    _Float16* __restrict__ og) {       // [B,N,C] f16
  __shared__ _Float16 Kbuf[2][SLAB * HEAD_D];   // [key][d], 2 x 8 KB
  __shared__ _Float16 Vt[2][HEAD_D * SLAB];     // [d][key], 2 x 8 KB
  __shared__ _Float16 Pst[8][16 * SLAB];        // per-wave P staging, 16 KB

  const int wave = threadIdx.x >> 5;
  const int lane = threadIdx.x & 31;
  const int half = lane >> 4;
  const int nl = lane & 15;

  const int bh = blockIdx.x >> 4;            // 0..31 = b*8 + head
  const int qblk = blockIdx.x & 15;          // 16 blocks of 128 queries
  const int b = bh >> 3;
  const int m0 = qblk * 128 + wave * 16;     // query tile start (per head)

  const _Float16* Qb = qg + (size_t)bh * N_TOK * HEAD_D;
  const _Float16* Kb = kg + (size_t)bh * N_TOK * HEAD_D;
  const _Float16* Vb = vg + (size_t)bh * N_TOK * HEAD_D;
  const float* biasb = biasg + (size_t)b * N_TOK * N_TOK;
  const float beta = beta_p[0];

  // LDS byte offsets (flat-address low 32 bits == LDS offset per ISA 10.2)
  const unsigned klds0 = (unsigned)(uintptr_t)&Kbuf[0][0];
  const unsigned klds1 = (unsigned)(uintptr_t)&Kbuf[1][0];

  // Q tile 16x64, two K-chunks of 32 (held in registers for all slabs)
  v16h qa0 = load_a_frag(Qb, HEAD_D, m0, 0, lane);
  v16h qa1 = load_a_frag(Qb, HEAD_D, m0, 32, lane);

  // V prefetch assignment: thread covers (key row, 16 d's) of the slab
  const int vkey = threadIdx.x >> 2;          // 0..63
  const int vdc = (threadIdx.x & 3) * 16;     // 0,16,32,48

  // ---- prologue: slab 0 in flight ----
  async_copy_slab(klds0, Kb, wave, lane);
  v8h vr0 = *(const v8h*)(Vb + (size_t)vkey * HEAD_D + vdc);
  v8h vr1 = *(const v8h*)(Vb + (size_t)vkey * HEAD_D + vdc + 8);

  v8f o[4] = {};
  float mrow[8], lrow[8];
#pragma unroll
  for (int r = 0; r < 8; ++r) { mrow[r] = -1e30f; lrow[r] = 0.0f; }

  int buf = 0;
  for (int slab = 0; slab < NSLAB; ++slab) {
    const int kt = slab * SLAB;
    // store prefetched V registers transposed into Vt[buf]
#pragma unroll
    for (int e = 0; e < 8; ++e) {
      Vt[buf][(vdc + e) * SLAB + vkey] = vr0[e];
      Vt[buf][(vdc + 8 + e) * SLAB + vkey] = vr1[e];
    }
    // prefetch slab+1 (K via async DMA into the other buffer, V into regs)
    if (slab + 1 < NSLAB) {
      const _Float16* Kn = Kb + (size_t)(kt + SLAB) * HEAD_D;
      async_copy_slab(buf ? klds0 : klds1, Kn, wave, lane);
      const _Float16* Vn = Vb + (size_t)(kt + SLAB) * HEAD_D;
      vr0 = *(const v8h*)(Vn + (size_t)vkey * HEAD_D + vdc);
      vr1 = *(const v8h*)(Vn + (size_t)vkey * HEAD_D + vdc + 8);
      // drain this slab's 2 async copies; leave the 2 newest in flight
      asm volatile("s_wait_asynccnt 0x2" ::: "memory");
    } else {
      asm volatile("s_wait_asynccnt 0x0" ::: "memory");
    }
    __syncthreads();   // all waves' K-DMA + V-stores for this slab visible

    // ---- scores: four 16-key subtiles, K-dim = 64 (2 wmmas each) ----
    v8f s[4];
#pragma unroll
    for (int st = 0; st < 4; ++st) {
      v8f c = {};
      {
        const _Float16* p = &Kbuf[buf][(st * 16 + nl) * HEAD_D + half * 16];
        c = wmma16x16x32(qa0, pack16(*(const v8h*)p, *(const v8h*)(p + 8)), c);
      }
      {
        const _Float16* p =
            &Kbuf[buf][(st * 16 + nl) * HEAD_D + 32 + half * 16];
        c = wmma16x16x32(qa1, pack16(*(const v8h*)p, *(const v8h*)(p + 8)), c);
      }
      s[st] = c;
    }

    // ---- online softmax over this 64-key slab ----
#pragma unroll
    for (int r = 0; r < 8; ++r) {
      const int qrow = m0 + r + half * 8;
      const float* bp = biasb + (size_t)qrow * N_TOK + kt;
      float sv[4];
      float mx = -1e30f;
#pragma unroll
      for (int st = 0; st < 4; ++st) {
        sv[st] = s[st][r] * 0.125f + beta * bp[st * 16 + nl];
        mx = fmaxf(mx, sv[st]);
      }
      mx = fmaxf(mx, __shfl_xor(mx, 1, 32));
      mx = fmaxf(mx, __shfl_xor(mx, 2, 32));
      mx = fmaxf(mx, __shfl_xor(mx, 4, 32));
      mx = fmaxf(mx, __shfl_xor(mx, 8, 32));
      const float nm = fmaxf(mrow[r], mx);
      float ps = 0.0f;
#pragma unroll
      for (int st = 0; st < 4; ++st) {
        const float pv = __expf(sv[st] - nm);
        Pst[wave][(r + half * 8) * SLAB + st * 16 + nl] = (_Float16)pv;
        ps += pv;
      }
      ps += __shfl_xor(ps, 1, 32);
      ps += __shfl_xor(ps, 2, 32);
      ps += __shfl_xor(ps, 4, 32);
      ps += __shfl_xor(ps, 8, 32);
      const float corr = __expf(mrow[r] - nm);
      lrow[r] = lrow[r] * corr + ps;
      mrow[r] = nm;
#pragma unroll
      for (int t = 0; t < 4; ++t) o[t][r] *= corr;
    }
    // per-wave LDS RAW: DS ops are in-order per wave; fence the compiler and
    // drain the DS counter before the cross-lane A-layout read-back.
    asm volatile("s_wait_dscnt 0" ::: "memory");

    // ---- P (16x64) x V (64x64): 2 A-frags x 4 d-tiles = 8 wmmas ----
    v16h pa0, pa1;
    {
      const _Float16* p = &Pst[wave][nl * SLAB + half * 16];
      pa0 = pack16(*(const v8h*)p, *(const v8h*)(p + 8));
      pa1 = pack16(*(const v8h*)(p + 32), *(const v8h*)(p + 40));
    }
#pragma unroll
    for (int t = 0; t < 4; ++t) {
      const _Float16* p0 = &Vt[buf][(t * 16 + nl) * SLAB + half * 16];
      o[t] = wmma16x16x32(pa0, pack16(*(const v8h*)p0, *(const v8h*)(p0 + 8)),
                          o[t]);
      const _Float16* p1 = &Vt[buf][(t * 16 + nl) * SLAB + 32 + half * 16];
      o[t] = wmma16x16x32(pa1, pack16(*(const v8h*)p1, *(const v8h*)(p1 + 8)),
                          o[t]);
    }
    __syncthreads();   // all reads of buf done before it is overwritten
    buf ^= 1;
  }

  // ---- epilogue: normalize, write [B,N,C] f16 ----
  const int head = bh & (HEADS - 1);
#pragma unroll
  for (int t = 0; t < 4; ++t) {
#pragma unroll
    for (int r = 0; r < 8; ++r) {
      const int qrow = m0 + r + half * 8;
      const size_t token = (size_t)b * N_TOK + qrow;
      const int col = head * HEAD_D + t * 16 + nl;
      og[token * C_DIM + col] = (_Float16)(o[t][r] / lrow[r]);
    }
  }
}

// ---------------------------------------------------------------------------
// Kernel 5: output projection + bias + residual (fp32 out). 32x64 tiles.
// ---------------------------------------------------------------------------
__global__ __launch_bounds__(256)
void sa_outproj_kernel(const _Float16* __restrict__ a,   // [8192][512] f16
                       const _Float16* __restrict__ w,   // [512][512] f16
                       const float* __restrict__ ob,
                       const float* __restrict__ x,
                       float* __restrict__ out) {
  const int wave = (blockIdx.x * blockDim.x + threadIdx.x) >> 5;
  const int lane = threadIdx.x & 31;
  const int nb = wave & 7;          // 8 N-blocks of 64
  const int mt = wave >> 3;         // 256 M-tiles of 32
  const int m0 = mt * 32;
  const int n0 = nb * 64;

  v8f acc[2][4] = {};
  for (int k0 = 0; k0 < C_DIM; k0 += 32) {
    const v16h a0 = load_a_frag(a, C_DIM, m0, k0, lane);
    const v16h a1 = load_a_frag(a, C_DIM, m0 + 16, k0, lane);
#pragma unroll
    for (int t = 0; t < 4; ++t) {
      const v16h bf = load_bt_frag(w, C_DIM, n0 + t * 16, k0, lane);
      acc[0][t] = wmma16x16x32(a0, bf, acc[0][t]);
      acc[1][t] = wmma16x16x32(a1, bf, acc[1][t]);
    }
  }

  const int half = lane >> 4;
  const int nl = lane & 15;
#pragma unroll
  for (int t = 0; t < 4; ++t) {
    const int j = n0 + t * 16 + nl;
    const float bj = ob[j];
#pragma unroll
    for (int rt = 0; rt < 2; ++rt) {
#pragma unroll
      for (int r = 0; r < 8; ++r) {
        const int tok = m0 + rt * 16 + r + half * 8;
        const size_t idx = (size_t)tok * C_DIM + j;
        out[idx] = x[idx] + acc[rt][t][r] + bj;
      }
    }
  }
}

// ---------------------------------------------------------------------------
// Host launcher
// ---------------------------------------------------------------------------
extern "C" void kernel_launch(void* const* d_in, const int* in_sizes, int n_in,
                              void* d_out, int out_size, void* d_ws, size_t ws_size,
                              hipStream_t stream) {
  const float* x      = (const float*)d_in[0];
  const float* bias   = (const float*)d_in[1];
  const float* qkv_w  = (const float*)d_in[2];
  const float* qkv_b  = (const float*)d_in[3];
  const float* out_w  = (const float*)d_in[4];
  const float* out_b  = (const float*)d_in[5];
  const float* ln_g   = (const float*)d_in[6];
  const float* ln_b   = (const float*)d_in[7];
  const float* beta   = (const float*)d_in[8];
  float* out = (float*)d_out;

  char* ws = (char*)d_ws;
  size_t off = 0;
  auto walloc = [&](size_t bytes) -> void* {
    off = (off + 255) & ~(size_t)255;
    void* p = ws + off;
    off += bytes;
    return p;
  };
  _Float16* h16    = (_Float16*)walloc((size_t)ROWS * C_DIM * 2);      // 8 MB
  _Float16* qkvw16 = (_Float16*)walloc((size_t)QKV_DIM * C_DIM * 2);   // 1.5 MB
  _Float16* outw16 = (_Float16*)walloc((size_t)C_DIM * C_DIM * 2);     // 0.5 MB
  _Float16* q16    = (_Float16*)walloc((size_t)ROWS * C_DIM * 2);      // 8 MB
  _Float16* k16    = (_Float16*)walloc((size_t)ROWS * C_DIM * 2);      // 8 MB
  _Float16* v16    = (_Float16*)walloc((size_t)ROWS * C_DIM * 2);      // 8 MB
  _Float16* att16  = (_Float16*)walloc((size_t)ROWS * C_DIM * 2);      // 8 MB
  (void)ws_size; (void)n_in; (void)in_sizes; (void)out_size;

  // 1) LayerNorm -> f16
  sa_ln_kernel<<<ROWS, 256, 0, stream>>>(x, ln_g, ln_b, h16);

  // 2) weight conversion
  sa_cvt_f16_kernel<<<(QKV_DIM * C_DIM + 255) / 256, 256, 0, stream>>>(
      qkv_w, qkvw16, QKV_DIM * C_DIM);
  sa_cvt_f16_kernel<<<(C_DIM * C_DIM + 255) / 256, 256, 0, stream>>>(
      out_w, outw16, C_DIM * C_DIM);

  // 3) QKV GEMM: 256 M-tiles x 24 N-blocks = 6144 waves / 8 per block
  sa_qkv_gemm_kernel<<<6144 / 8, 256, 0, stream>>>(h16, qkvw16, qkv_b,
                                                   q16, k16, v16);

  // 4) attention: 32 (b,h) x 16 query-blocks of 128
  sa_attn_kernel<<<B_DIM * HEADS * 16, 256, 0, stream>>>(q16, k16, v16, bias,
                                                         beta, att16);

  // 5) out projection + residual: 256 M-tiles x 8 N-blocks = 2048 waves
  sa_outproj_kernel<<<2048 / 8, 256, 0, stream>>>(att16, outw16, out_b, x, out);
}
